// CrossAttentionFusion_Reverse_89593017795354
// MI455X (gfx1250) — matrix operand order repaired and checked by reading, not compiled, not run on hardware
//
#include <hip/hip_runtime.h>
#include <cstddef>

typedef __attribute__((ext_vector_type(16))) __bf16 v16bf;
typedef __attribute__((ext_vector_type(8)))  __bf16 v8bf;
typedef __attribute__((ext_vector_type(8)))  float  v8f;
typedef __attribute__((ext_vector_type(4)))  float  v4f;

#define BB   4
#define CC   320
#define DD   16
#define HHs  32
#define WWs  32
#define SS   16384
#define NHT  32          // HEADS * TOK
#define QDIM 1280
#define EPSI 1e-5f
#define ATT_SCALE 0.15811388300841897f   // 40^-0.5

static __device__ __forceinline__ float  bf2f(__bf16 x) { return (float)x; }
static __device__ __forceinline__ __bf16 f2bf(float x)  { return (__bf16)x; }

static __device__ __forceinline__ v8f wmma_bf16(v16bf a, v16bf b, v8f c) {
  return __builtin_amdgcn_wmma_f32_16x16x32_bf16(false, a, false, b, (short)0, c, false, false);
}

// ---------------- prep: [B,C,S] f32 -> [B,S,C] bf16 ----------------
__global__ void k_transpose(const float* __restrict__ img, __bf16* __restrict__ xt) {
  int g = blockIdx.x * 256 + threadIdx.x;          // g = b*SS + s
  int b = g >> 14; int s = g & (SS - 1);
  const float* src = img + (size_t)b * CC * SS + s;
  __bf16* dst = xt + (size_t)g * CC;
  for (int cg = 0; cg < CC / 8; ++cg) {
    v8bf o;
#pragma unroll
    for (int j = 0; j < 8; ++j) o[j] = f2bf(src[(size_t)(cg * 8 + j) * SS]);
    *(v8bf*)(dst + cg * 8) = o;
  }
}

// -------- pack conv weights into WMMA A-fragment layout (bf16) -----
// wp[((mt*KT + kt)*32 + lane)*16 + e],  K_local = e + 16*hi
__global__ void k_pack_wf(const float* __restrict__ Wf, __bf16* __restrict__ wp,
                          int cin, int ktTotal) {
  int tid = blockIdx.x * 256 + threadIdx.x;
  int total = 20 * ktTotal * 32;
  if (tid >= total) return;
  int lane = tid & 31;
  int kt = (tid >> 5) % ktTotal;
  int mt = tid / (32 * ktTotal);
  int m = mt * 16 + (lane & 15);
  int hi = lane >> 4;
  __bf16* dst = wp + (size_t)tid * 16;
#pragma unroll
  for (int e = 0; e < 16; ++e) {
    int Kl = e + 16 * hi;
    int Kg = kt * 32 + Kl;
    int tap = Kg / cin;
    int ci  = Kg % cin;
    dst[e] = f2bf(Wf[((size_t)m * cin + ci) * 27 + tap]);
  }
}

// ---------------- q = cli @ Wq^T + bq ----------------
__global__ void k_qlin(const float* __restrict__ cli, const float* __restrict__ Wq,
                       const float* __restrict__ bq, float* __restrict__ qlin) {
  int tid = blockIdx.x * 256 + threadIdx.x;
  if (tid >= BB * QDIM) return;
  int b = tid / QDIM, i = tid % QDIM;
  float acc = bq[i];
  for (int c = 0; c < CC; ++c) acc += cli[b * CC + c] * Wq[(size_t)i * CC + c];
  qlin[tid] = acc;
}

// ---------------- generic LayerNorm (one block per batch row) ------
__global__ void k_ln(const float* __restrict__ x, const float* __restrict__ w,
                     const float* __restrict__ bsh, float* __restrict__ y,
                     __bf16* __restrict__ ybf, int N) {
  __shared__ float red[256];
  int b = blockIdx.x, t = threadIdx.x;
  const float* xb = x + (size_t)b * N;
  float s = 0.f;
  for (int i = t; i < N; i += 256) s += xb[i];
  red[t] = s; __syncthreads();
  for (int off = 128; off; off >>= 1) { if (t < off) red[t] += red[t + off]; __syncthreads(); }
  float mu = red[0] / N;
  __syncthreads();
  float s2 = 0.f;
  for (int i = t; i < N; i += 256) { float d = xb[i] - mu; s2 += d * d; }
  red[t] = s2; __syncthreads();
  for (int off = 128; off; off >>= 1) { if (t < off) red[t] += red[t + off]; __syncthreads(); }
  float rs = rsqrtf(red[0] / N + EPSI);
  for (int i = t; i < N; i += 256) {
    float v = (xb[i] - mu) * rs * w[i] + bsh[i];
    y[(size_t)b * N + i] = v;
    if (ybf) ybf[(size_t)b * N + i] = f2bf(v);
  }
}

// -------- weff = scale * (q^T Wk), packed directly as A-fragments --
__global__ void k_weffpack(const float* __restrict__ qn, const float* __restrict__ Wk,
                           __bf16* __restrict__ wp) {
  int tid = blockIdx.x * 256 + threadIdx.x;
  if (tid >= BB * 640) return;                       // 4 * (2 mt * 10 kt * 32 lanes)
  int b = tid / 640, r = tid % 640;
  int mt = r / 320, r2 = r % 320;
  int kt = r2 / 32, lane = r2 % 32;
  int ht = mt * 16 + (lane & 15);
  int hi = lane >> 4;
  int h = ht >> 2, tt = ht & 3;
  const float* qb = qn + (size_t)b * QDIM + h * 160 + tt;
  __bf16* dst = wp + (size_t)tid * 16;
#pragma unroll
  for (int e = 0; e < 16; ++e) {
    int Kl = e + 16 * hi;
    int c = kt * 32 + Kl;
    float acc = 0.f;
    for (int d = 0; d < 40; ++d) acc += qb[d * 4] * Wk[(size_t)(h * 40 + d) * CC + c];
    dst[e] = f2bf(acc * ATT_SCALE);
  }
}

__global__ void k_c0(const float* __restrict__ qn, const float* __restrict__ bk,
                     float* __restrict__ c0) {
  int tid = threadIdx.x;  // 128
  int b = tid / NHT, ht = tid % NHT;
  int h = ht >> 2, tt = ht & 3;
  float acc = 0.f;
  for (int d = 0; d < 40; ++d) acc += qn[(size_t)b * QDIM + h * 160 + d * 4 + tt] * bk[h * 40 + d];
  c0[tid] = acc * ATT_SCALE;
}

// -------- scores[b,ht,s] = weff . x_t + c0   (WMMA GEMM) ----------
__global__ void k_scores(const __bf16* __restrict__ xt, const __bf16* __restrict__ weffp,
                         const float* __restrict__ c0, float* __restrict__ scores) {
  int lane = threadIdx.x & 31, wv = threadIdx.x >> 5;
  int id = blockIdx.x * 8 + wv;                 // 0..8191
  int mt = id & 1; int nt = (id >> 1) & 1023; int b = id >> 11;
  int n = lane & 15, hi = lane >> 4;
  int s0 = nt * 16;
  const __bf16* bbase = xt + ((size_t)(b * SS + s0 + n)) * CC + hi * 16;
  const __bf16* abase = weffp + ((size_t)((b * 20 + mt * 10) * 32 + lane)) * 16;
  v8f acc;
  int cb0 = b * NHT + mt * 16 + 8 * hi;
#pragma unroll
  for (int r = 0; r < 8; ++r) acc[r] = c0[cb0 + r];
#pragma unroll
  for (int kt = 0; kt < 10; ++kt) {
    v16bf a  = *(const v16bf*)(abase + (size_t)kt * 512);
    v16bf bv = *(const v16bf*)(bbase + kt * 32);
    acc = wmma_bf16(a, bv, acc);
  }
  float* out = scores + ((size_t)(b * NHT + mt * 16 + 8 * hi)) * SS + s0 + n;
#pragma unroll
  for (int r = 0; r < 8; ++r) out[(size_t)r * SS] = acc[r];
}

// ---------------- softmax over S, write bf16 ----------------
__global__ void k_softmax(const float* __restrict__ scores, __bf16* __restrict__ attn) {
  __shared__ float red[256];
  int row = blockIdx.x, t = threadIdx.x;
  const float* x = scores + (size_t)row * SS;
  float mx = -1e30f;
  for (int i = t; i < SS; i += 256) mx = fmaxf(mx, x[i]);
  red[t] = mx; __syncthreads();
  for (int off = 128; off; off >>= 1) { if (t < off) red[t] = fmaxf(red[t], red[t + off]); __syncthreads(); }
  mx = red[0];
  __syncthreads();
  float s = 0.f;
  for (int i = t; i < SS; i += 256) s += expf(x[i] - mx);
  red[t] = s; __syncthreads();
  for (int off = 128; off; off >>= 1) { if (t < off) red[t] += red[t + off]; __syncthreads(); }
  float inv = 1.f / red[0];
  __bf16* o = attn + (size_t)row * SS;
  for (int i = t; i < SS; i += 256) o[i] = f2bf(expf(x[i] - mx) * inv);
}

// -------- G[b,c,ht] = sum_s x[c,s]*attn[ht,s]  (split-K WMMA + atomics)
__global__ void k_gmat(const float* __restrict__ img, const __bf16* __restrict__ attn,
                       float* __restrict__ G) {
  int lane = threadIdx.x & 31, wv = threadIdx.x >> 5;
  int id = blockIdx.x * 8 + wv;                 // 0..5119
  int kseg = id & 31; int nt = (id >> 5) & 1; int mt = (id >> 6) % 20; int b = id / 1280;
  int nl = lane & 15, hi = lane >> 4;
  int c = mt * 16 + nl;                         // A-row
  const float* arow = img + ((size_t)(b * CC + c)) * SS;
  const __bf16* brow = attn + ((size_t)(b * NHT + nt * 16 + nl)) * SS + hi * 16;
  v8f acc;
#pragma unroll
  for (int r = 0; r < 8; ++r) acc[r] = 0.f;
  int k0base = kseg * 512;
  for (int kc = 0; kc < 16; ++kc) {
    int k0 = k0base + kc * 32;
    v4f f0 = *(const v4f*)(arow + k0 + 16 * hi);
    v4f f1 = *(const v4f*)(arow + k0 + 16 * hi + 4);
    v4f f2 = *(const v4f*)(arow + k0 + 16 * hi + 8);
    v4f f3 = *(const v4f*)(arow + k0 + 16 * hi + 12);
    v16bf a;
#pragma unroll
    for (int j = 0; j < 4; ++j) {
      a[j]      = f2bf(f0[j]);
      a[4 + j]  = f2bf(f1[j]);
      a[8 + j]  = f2bf(f2[j]);
      a[12 + j] = f2bf(f3[j]);
    }
    v16bf bv = *(const v16bf*)(brow + k0);
    acc = wmma_bf16(a, bv, acc);
  }
  float* g = G + ((size_t)(b * CC + mt * 16 + 8 * hi)) * NHT + nt * 16 + nl;
#pragma unroll
  for (int r = 0; r < 8; ++r) atomicAdd(g + (size_t)r * NHT, acc[r]);
}

// -------- o_flat = Wv_head . G + bv ----------
__global__ void k_t1(const float* __restrict__ G, const float* __restrict__ Wv,
                     const float* __restrict__ bv, float* __restrict__ t1) {
  int tid = blockIdx.x * 256 + threadIdx.x;
  if (tid >= BB * QDIM) return;
  int b = tid / QDIM, i = tid % QDIM;
  int h = i / 160, d = (i % 160) / 4, tt = i % 4;
  int oc = h * 40 + d, ht = h * 4 + tt;
  float acc = bv[oc];
  for (int c = 0; c < CC; ++c) acc += Wv[(size_t)oc * CC + c] * G[((size_t)(b * CC + c)) * NHT + ht];
  t1[tid] = acc;
}

__global__ void k_t2(const float* __restrict__ t1, const float* __restrict__ Wo,
                     const float* __restrict__ bo, float* __restrict__ t2) {
  int tid = blockIdx.x * 256 + threadIdx.x;
  if (tid >= BB * CC) return;
  int b = tid / CC, j = tid % CC;
  float acc = bo[j];
  for (int i = 0; i < QDIM; ++i) acc += t1[(size_t)b * QDIM + i] * Wo[(size_t)j * QDIM + i];
  t2[tid] = acc;
}

// ---------------- implicit-GEMM 3x3x3 conv with WMMA ----------------
// 2x2 register-blocked: each wave computes 32 Cout x 32 positions (4 wmma / K-chunk,
// A/B fragments each reused twice -> 2 b128 loads per wmma instead of 4).
// OUTM 0: bf16 [S,C] store; 1: f32 [C,S] store
template <int CBN, int OUTM>
__global__ void k_conv(const __bf16* __restrict__ xt, const __bf16* __restrict__ bcast,
                       const __bf16* __restrict__ wp, const float* __restrict__ bias,
                       void* __restrict__ outp) {
  constexpr int KT = 27 * CBN;
  int lane = threadIdx.x & 31, wv = threadIdx.x >> 5;
  int id = blockIdx.x * 8 + wv;                 // 0..20479
  int mp = id % 10; int nt = (id / 10) & 511; int b = id / 5120;
  int n = lane & 15, hi = lane >> 4;
  int s0 = nt * 32;                             // 32 positions: one full W-row segment
  int h0 = (s0 >> 5) & 31, d0 = s0 >> 10;      // uniform over all 32 positions
  const __bf16* xb = xt + ((size_t)b * SS) * CC;
  const __bf16* bc = bcast ? bcast + (size_t)b * CC : nullptr;
  const __bf16* wbase0 = wp + (size_t)(mp * 2) * KT * 512 + (size_t)lane * 16;
  const __bf16* wbase1 = wbase0 + (size_t)KT * 512;
  v8f a00, a01, a10, a11;                       // acc[Mtile][Ntile]
#pragma unroll
  for (int r = 0; r < 8; ++r) { a00[r] = 0.f; a01[r] = 0.f; a10[r] = 0.f; a11[r] = 0.f; }
  v16bf zv;
#pragma unroll
  for (int e = 0; e < 16; ++e) zv[e] = (__bf16)0.0f;

  for (int tap = 0; tap < 27; ++tap) {
    int kd = tap / 9 - 1, kh = (tap / 3) % 3 - 1, kw = tap % 3 - 1;
    int dd = d0 + kd, hh = h0 + kh;
    if (dd < 0 || dd >= DD || hh < 0 || hh >= HHs) continue;   // wave-uniform
    int ww0 = n + kw;                  // N-subtile 0: w = n
    int ww1 = 16 + n + kw;             // N-subtile 1: w = 16 + n
    bool inb0 = (ww0 >= 0) && (ww0 < WWs);
    bool inb1 = (ww1 >= 0) && (ww1 < WWs);
    const __bf16* bpos0 = xb + ((size_t)(s0 + n + kd * 1024 + kh * 32 + kw)) * CC + hi * 16;
    const __bf16* bpos1 = bpos0 + (size_t)16 * CC;
    const __bf16* wk0 = wbase0 + (size_t)tap * CBN * 512;
    const __bf16* wk1 = wbase1 + (size_t)tap * CBN * 512;
#pragma unroll
    for (int cb = 0; cb < CBN; ++cb) {
      v16bf af0 = *(const v16bf*)(wk0 + (size_t)cb * 512);
      v16bf af1 = *(const v16bf*)(wk1 + (size_t)cb * 512);
      int cin0 = cb * 32;
      v16bf b0, b1;
      if (cin0 < CC) {
        b0 = inb0 ? *(const v16bf*)(bpos0 + cin0) : zv;
        b1 = inb1 ? *(const v16bf*)(bpos1 + cin0) : zv;
      } else {
        v16bf bcv = *(const v16bf*)(bc + (cin0 - CC) + hi * 16);
        b0 = inb0 ? bcv : zv;
        b1 = inb1 ? bcv : zv;
      }
      a00 = wmma_bf16(af0, b0, a00);
      a01 = wmma_bf16(af0, b1, a01);
      a10 = wmma_bf16(af1, b0, a10);
      a11 = wmma_bf16(af1, b1, a11);
    }
  }
#pragma unroll
  for (int mi = 0; mi < 2; ++mi) {
    int chb = (mp * 2 + mi) * 16 + 8 * hi;
#pragma unroll
    for (int ni = 0; ni < 2; ++ni) {
      v8f acc = (mi == 0) ? (ni == 0 ? a00 : a01) : (ni == 0 ? a10 : a11);
      int pos = s0 + ni * 16 + n;
      if (OUTM == 0) {
        v8bf o;
#pragma unroll
        for (int r = 0; r < 8; ++r) o[r] = f2bf(acc[r] + bias[chb + r]);
        *(v8bf*)((__bf16*)outp + ((size_t)(b * SS) + pos) * CC + chb) = o;
      } else {
        float* ob = (float*)outp + ((size_t)b * CC) * SS;
#pragma unroll
        for (int r = 0; r < 8; ++r) ob[(size_t)(chb + r) * SS + pos] = acc[r] + bias[chb + r];
      }
    }
  }
}

// -------- instance-norm stats over [B,S,C] bf16 (partial + atomics) --------
__global__ void k_stats1p(const __bf16* __restrict__ y1t, float* __restrict__ sum1,
                          float* __restrict__ sq1) {
  __shared__ float L1[256], L2[256];
  int bid = blockIdx.x, t = threadIdx.x;
  int sseg = bid & 7; int cg = (bid >> 3) % 20; int b = bid / 160;
  int chl = t & 15; int j = t >> 4;
  int ch = cg * 16 + chl;
  const __bf16* base = y1t + ((size_t)(b * SS + sseg * 2048 + j)) * CC + ch;
  float s1 = 0.f, s2 = 0.f;
  for (int i = 0; i < 128; ++i) {
    float v = bf2f(base[(size_t)i * 16 * CC]);
    s1 += v; s2 += v * v;
  }
  L1[t] = s1; L2[t] = s2; __syncthreads();
  for (int off = 8; off >= 1; off >>= 1) {
    if (j < off) { L1[t] += L1[t + off * 16]; L2[t] += L2[t + off * 16]; }
    __syncthreads();
  }
  if (t < 16) {
    atomicAdd(&sum1[b * CC + cg * 16 + t], L1[t]);
    atomicAdd(&sq1[b * CC + cg * 16 + t], L2[t]);
  }
}

__global__ void k_statsfin(const float* __restrict__ sum, const float* __restrict__ sq,
                           float* __restrict__ mean, float* __restrict__ rstd) {
  int tid = blockIdx.x * 256 + threadIdx.x;
  if (tid >= BB * CC) return;
  float m = sum[tid] * (1.f / SS);
  float v = sq[tid] * (1.f / SS) - m * m;
  mean[tid] = m;
  rstd[tid] = rsqrtf(v + EPSI);
}

// -------- instance-norm + exact GELU, [B,S,C] bf16 -> bf16 --------
__global__ void k_normgelu(const __bf16* __restrict__ y1t, const float* __restrict__ mean,
                           const float* __restrict__ rstd, __bf16* __restrict__ x2t) {
  int tid = blockIdx.x * 256 + threadIdx.x;       // BB*SS*40
  int chunk = tid % 40; int bs = tid / 40; int b = bs >> 14;
  int c0i = chunk * 8;
  v8bf in = *(const v8bf*)(y1t + (size_t)bs * CC + c0i);
  v8bf o;
#pragma unroll
  for (int e = 0; e < 8; ++e) {
    int idx = b * CC + c0i + e;
    float v = (bf2f(in[e]) - mean[idx]) * rstd[idx];
    o[e] = f2bf(0.5f * v * (1.f + erff(v * 0.70710678118654752f)));
  }
  *(v8bf*)(x2t + (size_t)bs * CC + c0i) = o;
}

// -------- stats over [B,C,S] f32 rows (one block / row) --------
__global__ void k_stats2(const float* __restrict__ y2, float* __restrict__ mean,
                         float* __restrict__ rstd) {
  __shared__ float R1[256], R2[256];
  int row = blockIdx.x, t = threadIdx.x;
  const float* x = y2 + (size_t)row * SS;
  float s1 = 0.f, s2 = 0.f;
  for (int i = t; i < SS; i += 256) { float v = x[i]; s1 += v; s2 += v * v; }
  R1[t] = s1; R2[t] = s2; __syncthreads();
  for (int off = 128; off; off >>= 1) {
    if (t < off) { R1[t] += R1[t + off]; R2[t] += R2[t + off]; }
    __syncthreads();
  }
  if (t == 0) {
    float m = R1[0] * (1.f / SS);
    float v = R2[0] * (1.f / SS) - m * m;
    mean[row] = m;
    rstd[row] = rsqrtf(v + EPSI);
  }
}

// -------- final: out = inorm(conv2) + img  (in place on d_out) --------
__global__ void k_final(float* __restrict__ outp, const float* __restrict__ img,
                        const float* __restrict__ mean, const float* __restrict__ rstd) {
  int tid = blockIdx.x * 256 + threadIdx.x;
  size_t base = (size_t)tid * 4;
  int bc = (int)(base >> 14);
  v4f v = *(const v4f*)(outp + base);
  v4f im = *(const v4f*)(img + base);
  float m = mean[bc], r = rstd[bc];
#pragma unroll
  for (int i = 0; i < 4; ++i) v[i] = (v[i] - m) * r + im[i];
  *(v4f*)(outp + base) = v;
}

extern "C" void kernel_launch(void* const* d_in, const int* in_sizes, int n_in,
                              void* d_out, int out_size, void* d_ws, size_t ws_size,
                              hipStream_t stream) {
  (void)in_sizes; (void)n_in; (void)out_size; (void)ws_size;
  const float* img   = (const float*)d_in[0];
  const float* cli   = (const float*)d_in[1];
  const float* Wq    = (const float*)d_in[2];
  const float* bq    = (const float*)d_in[3];
  const float* lnq_w = (const float*)d_in[4];
  const float* lnq_b = (const float*)d_in[5];
  const float* Wk    = (const float*)d_in[6];
  const float* bk    = (const float*)d_in[7];
  const float* Wv    = (const float*)d_in[8];
  const float* bvv   = (const float*)d_in[9];
  const float* Wo    = (const float*)d_in[10];
  const float* bo    = (const float*)d_in[11];
  const float* lno_w = (const float*)d_in[12];
  const float* lno_b = (const float*)d_in[13];
  const float* Wf1   = (const float*)d_in[14];
  const float* bf1   = (const float*)d_in[15];
  const float* Wf2   = (const float*)d_in[16];
  const float* bf2   = (const float*)d_in[17];

  char* ws = (char*)d_ws;
  size_t off = 0;
  auto alloc = [&](size_t bytes) -> char* {
    char* p = ws + off;
    off += (bytes + 255) & ~(size_t)255;
    return p;
  };
  // zero-initialized region (atomic accumulators) first
  float* G    = (float*)alloc((size_t)BB * CC * NHT * 4);
  float* sum1 = (float*)alloc((size_t)BB * CC * 4);
  float* sq1  = (float*)alloc((size_t)BB * CC * 4);
  size_t zeroBytes = off;
  float* mean1 = (float*)alloc((size_t)BB * CC * 4);
  float* rstd1 = (float*)alloc((size_t)BB * CC * 4);
  float* mean2 = (float*)alloc((size_t)BB * CC * 4);
  float* rstd2 = (float*)alloc((size_t)BB * CC * 4);
  float* qlin  = (float*)alloc((size_t)BB * QDIM * 4);
  float* qn    = (float*)alloc((size_t)BB * QDIM * 4);
  float* c0    = (float*)alloc((size_t)BB * NHT * 4);
  __bf16* weffp = (__bf16*)alloc((size_t)BB * 640 * 16 * 2);
  float* t1    = (float*)alloc((size_t)BB * QDIM * 4);
  float* t2    = (float*)alloc((size_t)BB * CC * 4);
  float* oattf = (float*)alloc((size_t)BB * CC * 4);
  __bf16* oattb = (__bf16*)alloc((size_t)BB * CC * 2);
  float* scores = (float*)alloc((size_t)BB * NHT * SS * 4);
  __bf16* attn  = (__bf16*)alloc((size_t)BB * NHT * SS * 2);
  __bf16* xt    = (__bf16*)alloc((size_t)BB * SS * CC * 2);  // also reused as x2t
  __bf16* y1t   = (__bf16*)alloc((size_t)BB * SS * CC * 2);
  __bf16* wp1   = (__bf16*)alloc((size_t)20 * 540 * 512 * 2);
  __bf16* wp2   = (__bf16*)alloc((size_t)20 * 270 * 512 * 2);
  __bf16* x2t   = xt;   // x_t dead after conv1; alias

  hipMemsetAsync(d_ws, 0, zeroBytes, stream);

  // prep
  k_transpose<<<(BB * SS) / 256, 256, 0, stream>>>(img, xt);
  k_pack_wf<<<(20 * 540 * 32) / 256, 256, 0, stream>>>(Wf1, wp1, 640, 540);
  k_pack_wf<<<(20 * 270 * 32) / 256, 256, 0, stream>>>(Wf2, wp2, 320, 270);

  // query path
  k_qlin<<<(BB * QDIM) / 256, 256, 0, stream>>>(cli, Wq, bq, qlin);
  k_ln<<<BB, 256, 0, stream>>>(qlin, lnq_w, lnq_b, qn, nullptr, QDIM);
  k_weffpack<<<(BB * 640) / 256, 256, 0, stream>>>(qn, Wk, weffp);
  k_c0<<<1, 128, 0, stream>>>(qn, bk, c0);

  // attention (K/V folded away)
  k_scores<<<(BB * 2 * 1024) / 8, 256, 0, stream>>>(xt, weffp, c0, scores);
  k_softmax<<<BB * NHT, 256, 0, stream>>>(scores, attn);
  k_gmat<<<(BB * 20 * 2 * 32) / 8, 256, 0, stream>>>(img, attn, G);
  k_t1<<<(BB * QDIM) / 256, 256, 0, stream>>>(G, Wv, bvv, t1);
  k_t2<<<(BB * CC) / 256, 256, 0, stream>>>(t1, Wo, bo, t2);
  k_ln<<<BB, 256, 0, stream>>>(t2, lno_w, lno_b, oattf, oattb, CC);

  // fused conv pipeline (2x2 register-blocked implicit GEMM)
  k_conv<20, 0><<<(BB * 10 * 512) / 8, 256, 0, stream>>>(xt, oattb, wp1, bf1, (void*)y1t);
  k_stats1p<<<BB * 20 * 8, 256, 0, stream>>>(y1t, sum1, sq1);
  k_statsfin<<<(BB * CC + 255) / 256, 256, 0, stream>>>(sum1, sq1, mean1, rstd1);
  k_normgelu<<<(BB * SS * 40) / 256, 256, 0, stream>>>(y1t, mean1, rstd1, x2t);
  k_conv<10, 1><<<(BB * 10 * 512) / 8, 256, 0, stream>>>(x2t, nullptr, wp2, bf2, d_out);
  k_stats2<<<BB * CC, 256, 0, stream>>>((const float*)d_out, mean2, rstd2);
  k_final<<<(BB * CC * SS / 4) / 256, 256, 0, stream>>>((float*)d_out, img, mean2, rstd2);
}
